// IBiLSTM_36876589203694
// MI455X (gfx1250) — matrix operand order repaired and checked by reading, not compiled, not run on hardware
//
#include <hip/hip_runtime.h>
#include <hip/hip_bf16.h>

typedef __attribute__((ext_vector_type(16))) __bf16 v16bf;
typedef __attribute__((ext_vector_type(8)))  float  v8f;
typedef __attribute__((ext_vector_type(4)))  unsigned int v4u;
typedef __attribute__((ext_vector_type(8)))  int v8i;
typedef __attribute__((ext_vector_type(4)))  int v4i;

union FragU { v16bf v; uint4 q[2]; };
union XFrag { uint4 q; __bf16 e[8]; };

// ---------------- problem sizes ----------------
// B=64, T=512, IN=512, H=512, 4H=2048, BT=32768
#define NB   64
#define NT   512
#define NIN  512
#define NH   512
#define NG   2048
#define NBT  32768

// ---------------- workspace layout (bytes) ----------------
// xswz   : swizzled bf16 x, A-fragment layout   [mt=2048][kc=16][lane*16+e]  = 33,554,432
// wxswz  : swizzled bf16 Wx, B-frag layout      [d=2][kc=16][nt=128][512]    =  4,194,304
// whswz  : swizzled bf16 Wh, B-frag layout      [d=2][kc=16][nt=128][512]    =  4,194,304
// bsum   : f32 bx+bh                            [d=2][2048]                  =     16,384
// xproj  : bf16 x@Wx + biases                   [d=2][t=512][g=2048][b=64]   = 268,435,456
// hbuf   : bf16 h ping-pong                     [d=2][buf=2][64*512]         =    262,144
// ctrl   : unsigned barrier counters            [d=2] (+pad)                 =        256
static constexpr size_t SZ_XSWZ  = (size_t)NBT * NIN * 2;
static constexpr size_t SZ_W     = (size_t)2 * NIN * NG * 2;
static constexpr size_t OFF_XSWZ = 0;
static constexpr size_t OFF_WX   = OFF_XSWZ + SZ_XSWZ;
static constexpr size_t OFF_WH   = OFF_WX + SZ_W;
static constexpr size_t OFF_BS   = OFF_WH + SZ_W;
static constexpr size_t OFF_XPRJ = OFF_BS + 16384;
static constexpr size_t SZ_XPRJ  = (size_t)2 * NT * NB * NG * 2;
static constexpr size_t OFF_HBUF = OFF_XPRJ + SZ_XPRJ;
static constexpr size_t SZ_HBUF  = (size_t)2 * 2 * NB * NH * 2;
static constexpr size_t OFF_CTRL = OFF_HBUF + SZ_HBUF;

__device__ __forceinline__ float hsig(float x) {
    return fminf(fmaxf(fmaf(x, 0.2f, 0.5f), 0.0f), 1.0f);
}
__device__ __forceinline__ float ssgn(float x) { return x / (1.0f + fabsf(x)); }

// A-fragment element->K mapping (16-bit A 16x32, wave32):
// lane half lh=lane>>4: e<8 -> K = 8*lh+e ; e>=8 -> K = 16 + 8*lh + (e-8)
// B-fragment element->K mapping (16-bit B 32x16): same as A, N = lane&15

// ---------------- init: zero d_out, h buffers, barrier counters ----------------
__global__ void __launch_bounds__(256) k_init(float* __restrict__ out,
                                              __bf16* __restrict__ hbuf,
                                              unsigned* __restrict__ ctrl) {
    size_t tid = (size_t)blockIdx.x * blockDim.x + threadIdx.x;
    size_t nthreads = (size_t)gridDim.x * blockDim.x;
    const size_t nout = (size_t)NB * NT * NH;
    for (size_t i = tid; i < nout; i += nthreads) out[i] = 0.0f;
    const size_t nh = (size_t)2 * 2 * NB * NH;
    for (size_t i = tid; i < nh; i += nthreads) hbuf[i] = (__bf16)0.0f;
    if (tid < 8) ctrl[tid] = 0u;
}

// ---------------- swizzle x (f32 -> bf16 A-fragment layout) ----------------
__global__ void __launch_bounds__(256) k_swz_x(const float* __restrict__ x,
                                               __bf16* __restrict__ xswz) {
    size_t idx = (size_t)blockIdx.x * blockDim.x + threadIdx.x;  // 2,097,152 total
    if (idx >= (size_t)2048 * 16 * 32 * 2) return;
    int ehalf = (int)(idx & 1);
    int lane  = (int)((idx >> 1) & 31);
    int kc    = (int)((idx >> 6) & 15);
    int mt    = (int)(idx >> 10);
    int lh = lane >> 4, lc = lane & 15;
    int row = mt * 16 + lc;
    int k0  = kc * 32 + (ehalf ? (16 + 8 * lh) : (8 * lh));
    const float* src = x + (size_t)row * NIN + k0;        // 8 contiguous f32
    __bf16* dst = xswz + ((size_t)mt * 16 + kc) * 512 + lane * 16 + ehalf * 8;
#pragma unroll
    for (int e = 0; e < 8; ++e) dst[e] = (__bf16)src[e];
}

// ---------------- swizzle weights (f32 -> bf16 B-fragment layout) + bias sums ----------------
__global__ void __launch_bounds__(256) k_swz_w(const float* __restrict__ Wxf,
                                               const float* __restrict__ Whf,
                                               const float* __restrict__ Wxb,
                                               const float* __restrict__ Whb,
                                               const float* __restrict__ bxf,
                                               const float* __restrict__ bhf,
                                               const float* __restrict__ bxb,
                                               const float* __restrict__ bhb,
                                               __bf16* __restrict__ wxswz,
                                               __bf16* __restrict__ whswz,
                                               float* __restrict__ bsum) {
    size_t idx = (size_t)blockIdx.x * blockDim.x + threadIdx.x;  // 4,194,304 total
    if (idx < 4096) {
        int d = (int)(idx >> 11), n = (int)(idx & 2047);
        bsum[idx] = d ? (bxb[n] + bhb[n]) : (bxf[n] + bhf[n]);
    }
    if (idx >= (size_t)1 << 22) return;
    int e     = (int)(idx & 15);
    int lane  = (int)((idx >> 4) & 31);
    int nt    = (int)((idx >> 9) & 127);
    int kc    = (int)((idx >> 16) & 15);
    int which = (int)((idx >> 20) & 1);   // 0 = Wx, 1 = Wh
    int d     = (int)((idx >> 21) & 1);
    int k = kc * 32 + (e < 8 ? 8 * (lane >> 4) + e : 16 + 8 * (lane >> 4) + (e - 8));
    int n = nt * 16 + (lane & 15);
    const float* src = which ? (d ? Whb : Whf) : (d ? Wxb : Wxf);
    __bf16* dst = (which ? whswz : wxswz) + (size_t)d * (16 * 128 * 512)
                + ((size_t)kc * 128 + nt) * 512 + lane * 16 + e;
    *dst = (__bf16)src[(size_t)k * NG + n];
}

// ---------------- phase A: xproj[d] = x @ Wx[d] + bx + bh (bf16 WMMA, f32 accum) ----------------
// grid: (16, 256, 2) blocks of 256. WG tile 128x128, wave tile 32x64.
__global__ void __launch_bounds__(256) k_xproj(const __bf16* __restrict__ xswz,
                                               const __bf16* __restrict__ wxswz,
                                               const float* __restrict__ bsum,
                                               __bf16* __restrict__ xproj) {
    const int d  = blockIdx.z;
    const int tileN0 = blockIdx.x * 128;
    const int tid = threadIdx.x, lane = tid & 31, w = tid >> 5;
    const int wm = w & 3, wn = w >> 2;
    const int lh = lane >> 4, lc = lane & 15;
    const __bf16* wd = wxswz + (size_t)d * (16 * 128 * 512);
    const int mtg0 = blockIdx.y * 8 + wm * 2;
    const int ntg0 = blockIdx.x * 8 + wn * 4;

    v8f acc[2][4] = {};
#pragma unroll
    for (int kc = 0; kc < 16; ++kc) {
        FragU a[2], b[4];
#pragma unroll
        for (int i = 0; i < 2; ++i) {
            const uint4* src = (const uint4*)(xswz + ((size_t)(mtg0 + i) * 16 + kc) * 512 + lane * 16);
            a[i].q[0] = src[0]; a[i].q[1] = src[1];
        }
#pragma unroll
        for (int j = 0; j < 4; ++j) {
            const uint4* src = (const uint4*)(wd + ((size_t)kc * 128 + (ntg0 + j)) * 512 + lane * 16);
            b[j].q[0] = src[0]; b[j].q[1] = src[1];
        }
        if (kc + 1 < 16)
            __builtin_prefetch((const void*)(xswz + ((size_t)mtg0 * 16 + kc + 1) * 512 + lane * 16), 0, 1);
#pragma unroll
        for (int i = 0; i < 2; ++i)
#pragma unroll
            for (int j = 0; j < 4; ++j)
                acc[i][j] = __builtin_amdgcn_wmma_f32_16x16x32_bf16(
                    false, a[i].v, false, b[j].v, (short)0, acc[i][j], false, false);
    }
    // epilogue: add bias, store bf16 into [d][t][g][b] layout (batch innermost)
#pragma unroll
    for (int j = 0; j < 4; ++j) {
        int col = tileN0 + wn * 64 + j * 16 + lc;
        float bs = bsum[d * NG + col];
#pragma unroll
        for (int i = 0; i < 2; ++i) {
#pragma unroll
            for (int v = 0; v < 8; ++v) {
                int r = blockIdx.y * 128 + wm * 32 + i * 16 + v + 8 * lh; // row = b*T + t
                int bb = r >> 9, t = r & 511;
                float val = acc[i][j][v] + bs;
                xproj[(((size_t)d * NT + t) * NG + col) * NB + bb] = (__bf16)val;
            }
        }
    }
}

// ---------------- phase B: recurrence (persistent, 2 dirs x 16 WGs, Wh pinned in LDS) ------------
__global__ void __launch_bounds__(256) k_recur(const __bf16* __restrict__ whswz,
                                               const __bf16* __restrict__ xproj,
                                               __bf16* __restrict__ hbuf,
                                               float* __restrict__ out,
                                               unsigned* __restrict__ ctrl) {
    const int d = blockIdx.x >> 4;
    const int s = blockIdx.x & 15;           // owns hidden cols [s*32, s*32+32)
    const int tid = threadIdx.x, lane = tid & 31, w = tid >> 5;
    const int m = w & 3;                     // batch row block 16m..16m+15
    const int j = w >> 2;                    // 16-col half within the 32 owned cols
    const int lh = lane >> 4, lc = lane & 15;

    // Wh slice for this WG: 16 kc x 4 gates x 2 halves = 128 frags x 1KB = 128KB LDS
    __shared__ __align__(64) __bf16 ldsW[16 * 8 * 512];

    const __bf16* whd = whswz + (size_t)d * (16 * 128 * 512);

    // ---- TDM: one tensor_load_to_lds DMA pulls the whole 128KB strided slice ----
    // 3D tile, data_size=8B: X = 2048B contiguous (frag pair j2=0,1)
    //                        Y = 4 gates   @ stride 32KB
    //                        Z = 16 kc     @ stride 128KB
    // packed contiguously into LDS => ldsW frag order f = kc*8 + g*2 + j2.
    if (w == 0) {
        unsigned ldsAddr = (unsigned)(uintptr_t)(&ldsW[0]);
        unsigned long long ga = (unsigned long long)(uintptr_t)whd + (unsigned long long)s * 2048ull;
        v4u g0 = { 1u,                                    // count=1 (valid descriptor)
                   ldsAddr,                               // lds_addr
                   (unsigned)(ga & 0xFFFFFFFFull),        // global_addr[31:0]
                   (unsigned)((ga >> 32) & 0x1FFFFFFull) | (2u << 30) };  // addr[56:32] | type=2
        v8i g1 = { (int)(3u << 16),                       // data_size = 3 (8 bytes)
                   (int)(256u << 16),                     // tensor_dim0[15:0] = 256
                   (int)(4u << 16),                       // tensor_dim1[15:0] = 4
                   (int)(256u << 16),                     // tile_dim0 = 256
                   (int)(4u | (16u << 16)),               // tile_dim1 = 4, tile_dim2 = 16
                   (int)4096,                             // tensor_dim0_stride = 4096 (32KB)
                   (int)(16384u << 16),                   // tensor_dim1_stride[15:0] = 16384 (128KB)
                   0 };
        v4i g2 = { 16, 0, 0, 0 };                         // tensor_dim2 = 16
        v4i g3 = { 0, 0, 0, 0 };
        v8i g4 = { 0, 0, 0, 0, 0, 0, 0, 0 };              // unused trailing group (6-arg form)
        __builtin_amdgcn_tensor_load_to_lds(g0, g1, g2, g3, g4, 0);
        __builtin_amdgcn_s_wait_tensorcnt(0);
    }
    __syncthreads();

    const int colh = s * 32 + j * 16 + lc;   // hidden column this lane produces
    const int b0 = 16 * m + 8 * lh;          // first of the 8 batch rows this lane holds
    const size_t hstride = (size_t)NB * NH;
    v8f cst = {};

#pragma unroll 1
    for (int t = 0; t < NT; ++t) {
        const int tt = d ? (NT - 1 - t) : t;
        const __bf16* hin = hbuf + ((size_t)d * 2 + (t & 1)) * hstride;
        v8f acc0 = {}, acc1 = {}, acc2 = {}, acc3 = {};
#pragma unroll
        for (int kc = 0; kc < 16; ++kc) {
            FragU a;
            const __bf16* ap = hin + (size_t)(16 * m + lc) * NH + kc * 32 + 8 * lh;
            a.q[0] = *(const uint4*)(ap);
            a.q[1] = *(const uint4*)(ap + 16);
            const __bf16* wb = ldsW + (size_t)(kc * 8) * 512 + (size_t)j * 512 + lane * 16;
            acc0 = __builtin_amdgcn_wmma_f32_16x16x32_bf16(false, a.v, false,
                     *(const v16bf*)(wb + 0 * 1024), (short)0, acc0, false, false);
            acc1 = __builtin_amdgcn_wmma_f32_16x16x32_bf16(false, a.v, false,
                     *(const v16bf*)(wb + 1 * 1024), (short)0, acc1, false, false);
            acc2 = __builtin_amdgcn_wmma_f32_16x16x32_bf16(false, a.v, false,
                     *(const v16bf*)(wb + 2 * 1024), (short)0, acc2, false, false);
            acc3 = __builtin_amdgcn_wmma_f32_16x16x32_bf16(false, a.v, false,
                     *(const v16bf*)(wb + 3 * 1024), (short)0, acc3, false, false);
        }
        // gate pre-activations: one 16B load per gate (batch-contiguous layout)
        const __bf16* xp = xproj + ((size_t)d * NT + tt) * (size_t)NG * NB;
        XFrag xg[4];
#pragma unroll
        for (int g = 0; g < 4; ++g)
            xg[g].q = *(const uint4*)(xp + ((size_t)(g * NH + colh)) * NB + b0);

        __bf16* hout = hbuf + ((size_t)d * 2 + ((t + 1) & 1)) * hstride;
#pragma unroll
        for (int v = 0; v < 8; ++v) {
            int bb = b0 + v;
            float gi = acc0[v] + (float)xg[0].e[v];
            float gf = acc1[v] + (float)xg[1].e[v];
            float go = acc2[v] + (float)xg[2].e[v];
            float gc = acc3[v] + (float)xg[3].e[v];
            float it = hsig(gi), ft = hsig(gf), ot = hsig(go), ct = ssgn(gc);
            float cn = fmaf(ft, cst[v], it * ct);
            cst[v] = cn;
            float hn = ot * ssgn(cn);
            hout[(size_t)bb * NH + colh] = (__bf16)hn;
            atomicAdd(&out[((size_t)bb * NT + tt) * NH + colh], hn);
        }
        // grid barrier among the 16 WGs of this direction
        __threadfence();
        __syncthreads();
        if (tid == 0) {
            atomicAdd(&ctrl[d], 1u);
            unsigned tgt = 16u * (unsigned)(t + 1);
            while (*((volatile unsigned*)&ctrl[d]) < tgt) __builtin_amdgcn_s_sleep(2);
        }
        __syncthreads();
        __threadfence();
    }
}

// ---------------- final: out = hardsigmoid(out) ----------------
__global__ void __launch_bounds__(256) k_finish(float* __restrict__ out) {
    size_t tid = (size_t)blockIdx.x * blockDim.x + threadIdx.x;
    size_t nthreads = (size_t)gridDim.x * blockDim.x;
    const size_t n = (size_t)NB * NT * NH;
    for (size_t i = tid; i < n; i += nthreads) out[i] = hsig(out[i]);
}

extern "C" void kernel_launch(void* const* d_in, const int* in_sizes, int n_in,
                              void* d_out, int out_size, void* d_ws, size_t ws_size,
                              hipStream_t stream) {
    (void)in_sizes; (void)n_in; (void)out_size; (void)ws_size;
    const float* x   = (const float*)d_in[0];
    const float* Wxf = (const float*)d_in[1];
    const float* bxf = (const float*)d_in[2];
    const float* Whf = (const float*)d_in[3];
    const float* bhf = (const float*)d_in[4];
    const float* Wxb = (const float*)d_in[5];
    const float* bxb = (const float*)d_in[6];
    const float* Whb = (const float*)d_in[7];
    const float* bhb = (const float*)d_in[8];
    float* out = (float*)d_out;

    char* ws = (char*)d_ws;
    __bf16*   xswz  = (__bf16*)(ws + OFF_XSWZ);
    __bf16*   wxswz = (__bf16*)(ws + OFF_WX);
    __bf16*   whswz = (__bf16*)(ws + OFF_WH);
    float*    bsum  = (float*)(ws + OFF_BS);
    __bf16*   xproj = (__bf16*)(ws + OFF_XPRJ);
    __bf16*   hbuf  = (__bf16*)(ws + OFF_HBUF);
    unsigned* ctrl  = (unsigned*)(ws + OFF_CTRL);

    k_init<<<2048, 256, 0, stream>>>(out, hbuf, ctrl);
    k_swz_x<<<8192, 256, 0, stream>>>(x, xswz);
    k_swz_w<<<16384, 256, 0, stream>>>(Wxf, Whf, Wxb, Whb, bxf, bhf, bxb, bhb,
                                       wxswz, whswz, bsum);
    dim3 gA(16, 256, 2);
    k_xproj<<<gA, 256, 0, stream>>>(xswz, wxswz, bsum, xproj);
    k_recur<<<32, 256, 0, stream>>>(whswz, xproj, hbuf, out, ctrl);
    k_finish<<<2048, 256, 0, stream>>>(out);
}
